// Goggle_86122684219718
// MI455X (gfx1250) — matrix operand order repaired and checked by reading, not compiled, not run on hardware
//
#include <hip/hip_runtime.h>
#include <hip/hip_bf16.h>
#include <math.h>

// ---------------------------------------------------------------------------
// GOGGLE-style graph VAE forward for MI455X (gfx1250, wave32, WMMA).
// B=8192, D=256, ENC=256, DEC=64.
// ---------------------------------------------------------------------------

typedef __attribute__((ext_vector_type(16))) _Float16 v16h;
typedef __attribute__((ext_vector_type(8)))  float    v8f;
typedef __attribute__((ext_vector_type(2)))  _Float16 h2v;

#define NB   8192
#define ND   256
#define NENC 256
#define NDEC 64
#define LDT  264   // padded K-stride (halves) for K-major (transposed) B slabs

__device__ __forceinline__ float fast_tanhf(float x) {
#if __has_builtin(__builtin_amdgcn_tanhf)
  return __builtin_amdgcn_tanhf(x);      // v_tanh_f32 on gfx1250
#else
  return tanhf(x);
#endif
}

// ---- WMMA 16x16x32 f16 fragment loaders (gfx1250 wave32 layouts) ----------

// A: 16x32 tile from row-major f32, ld = row stride in elements.
// Lanes 0-15: row M=lane, K={0..7,16..23}; lanes 16-31: K={8..15,24..31}.
__device__ __forceinline__ v16h load_a_f32g(const float* __restrict__ src, int ld, int lane) {
  const int half = lane >> 4, r = lane & 15;
  v16h a;
#pragma unroll
  for (int v = 0; v < 8; ++v) {
    const int kb = ((v >> 2) << 4) + (half << 3) + ((v & 3) << 1);
    const float* p = src + r * ld + kb;
    a[2 * v]     = (_Float16)p[0];
    a[2 * v + 1] = (_Float16)p[1];
  }
  return a;
}

// A: 16x32 tile from row-major f16 (dword-pair loads -> b128 merges).
__device__ __forceinline__ v16h load_a_f16(const _Float16* __restrict__ src, int ld, int lane) {
  const int half = lane >> 4, r = lane & 15;
  v16h a;
#pragma unroll
  for (int v = 0; v < 8; ++v) {
    const int kb = ((v >> 2) << 4) + (half << 3) + ((v & 3) << 1);
    h2v p = *(const h2v*)(src + r * ld + kb);
    a[2 * v]     = p[0];
    a[2 * v + 1] = p[1];
  }
  return a;
}

// B: 32x16 tile from a K-major (transposed) f16 slab: base[n * ldk + k].
// Lanes 0-15: col N=lane, K={0..7,16..23}; lanes 16-31: col N=lane-16, K={8..15,24..31}.
__device__ __forceinline__ v16h load_b_T(const _Float16* base, int ldk, int n0, int k0, int lane) {
  const int half = lane >> 4, r = lane & 15;
  v16h b;
#pragma unroll
  for (int v = 0; v < 8; ++v) {
    const int kb = ((v >> 2) << 4) + (half << 3) + ((v & 3) << 1);
    h2v p = *(const h2v*)(base + (n0 + r) * ldk + k0 + kb);
    b[2 * v]     = p[0];
    b[2 * v + 1] = p[1];
  }
  return b;
}

// ---------------------------------------------------------------------------
// Kernel 1: adjacency  (sigmoid, forced self loops, threshold when iter>50)
// ---------------------------------------------------------------------------
__global__ __launch_bounds__(256) void k_adj(const float* __restrict__ gp,
                                             const int* __restrict__ iter,
                                             float* __restrict__ adj_out,
                                             float* __restrict__ adj_ws) {
  const int i = blockIdx.x, j = threadIdx.x;
  float g = 1.0f / (1.0f + __expf(-gp[i * ND + j]));
  if (i == j) g = 1.0f;
  if (iter[0] > 50 && !(g > 0.1f)) g = 0.0f;
  adj_out[i * ND + j] = g;
  adj_ws[i * ND + j]  = g;
}

// ---------------------------------------------------------------------------
// Kernel 2: structural degrees.  block 0 -> row sums (dout), block 1 -> col
// sums (din); clamped to >= 1.
// ---------------------------------------------------------------------------
__global__ __launch_bounds__(256) void k_deg(const float* __restrict__ adj,
                                             float* __restrict__ degs) {
  const int t = threadIdx.x;
  float c = 0.0f;
  if (blockIdx.x == 0) {
    for (int q = 0; q < ND; ++q) c += (adj[t * ND + q] != 0.0f) ? 1.0f : 0.0f;
  } else {
    for (int q = 0; q < ND; ++q) c += (adj[q * ND + t] != 0.0f) ? 1.0f : 0.0f;
  }
  degs[blockIdx.x * ND + t] = fmaxf(c, 1.0f);
}

// ---------------------------------------------------------------------------
// Kernel 3: normalized, transposed message matrix M (f16, L2-resident).
// M[i][j] = adj[j][i] / sqrt(dout[j] * din[i])
// ---------------------------------------------------------------------------
__global__ __launch_bounds__(256) void k_M(const float* __restrict__ adj,
                                           const float* __restrict__ degs,
                                           _Float16* __restrict__ Mh) {
  const int i = blockIdx.x, j = threadIdx.x;
  Mh[i * ND + j] = (_Float16)(adj[j * ND + i] * rsqrtf(degs[j] * degs[ND + i]));
}

// ---------------------------------------------------------------------------
// Kernel 4: encoder layer 1:  h = relu(x @ W_enc + b_enc), stored f16.
// Block: 256 rows x 64 cols, 8 waves; each wave owns two 16-row strips and
// shares every B fragment between them (2 WMMAs per ds_load pair).
// ---------------------------------------------------------------------------
__global__ __launch_bounds__(256) void k_enc1(const float* __restrict__ x,
                                              const float* __restrict__ Wenc,
                                              const float* __restrict__ benc,
                                              _Float16* __restrict__ hout) {
  __shared__ _Float16 WT[64 * LDT];     // W_enc slice, K-major, 33 KB
  const int t = threadIdx.x;
  const int n0g = blockIdx.y * 64;
  const int row0 = blockIdx.x * 256;

  for (int idx = t; idx < 64 * NENC; idx += 256) {
    const int k = idx >> 6, n = idx & 63;
    WT[n * LDT + k] = (_Float16)Wenc[k * NENC + n0g + n];
  }
  __syncthreads();

  const int wave = t >> 5, lane = t & 31;
  const int half = lane >> 4, r = lane & 15;
  const int rw0 = row0 + wave * 32;       // strip 0
  const int rw1 = rw0 + 16;               // strip 1

  v8f acc[2][4];
#pragma unroll
  for (int ii = 0; ii < 2; ++ii)
#pragma unroll
    for (int d = 0; d < 4; ++d)
#pragma unroll
      for (int v = 0; v < 8; ++v) acc[ii][d][v] = 0.0f;

  for (int kk = 0; kk < 8; ++kk) {
    const v16h a0 = load_a_f32g(x + rw0 * ND + kk * 32, ND, lane);
    const v16h a1 = load_a_f32g(x + rw1 * ND + kk * 32, ND, lane);
#pragma unroll
    for (int d = 0; d < 4; ++d) {
      const v16h b = load_b_T(WT, LDT, d * 16, kk * 32, lane);
      acc[0][d] = __builtin_amdgcn_wmma_f32_16x16x32_f16(false, a0, false, b,
                                                         (short)0, acc[0][d], false, false);
      acc[1][d] = __builtin_amdgcn_wmma_f32_16x16x32_f16(false, a1, false, b,
                                                         (short)0, acc[1][d], false, false);
    }
  }
#pragma unroll
  for (int ii = 0; ii < 2; ++ii) {
    const int rw = ii ? rw1 : rw0;
#pragma unroll
    for (int d = 0; d < 4; ++d) {
      const int n = n0g + d * 16 + r;
      const float bias = benc[n];
#pragma unroll
      for (int v = 0; v < 8; ++v) {
        const int m = rw + v + 8 * half;
        hout[m * NENC + n] = (_Float16)fmaxf(acc[ii][d][v] + bias, 0.0f);
      }
    }
  }
}

// ---------------------------------------------------------------------------
// Kernel 5: mu / logvar heads + reparameterized z (fused).
// Block: 128 rows x 32 cols; both heads share the A fragment per K-step.
// ---------------------------------------------------------------------------
__global__ __launch_bounds__(256) void k_enc2(const _Float16* __restrict__ hbuf,
                                              const float* __restrict__ Wmu,
                                              const float* __restrict__ bmu,
                                              const float* __restrict__ Wlv,
                                              const float* __restrict__ blv,
                                              const float* __restrict__ eps,
                                              float* __restrict__ mu_o,
                                              float* __restrict__ lv_o,
                                              float* __restrict__ z_o) {
  __shared__ _Float16 WmuT[32 * LDT];   // 16.5 KB
  __shared__ _Float16 WlvT[32 * LDT];   // 16.5 KB
  const int t = threadIdx.x;
  const int n0g = blockIdx.y * 32;
  const int row0 = blockIdx.x * 128;

  for (int idx = t; idx < 32 * NENC; idx += 256) {
    const int k = idx >> 5, n = idx & 31;
    WmuT[n * LDT + k] = (_Float16)Wmu[k * ND + n0g + n];
    WlvT[n * LDT + k] = (_Float16)Wlv[k * ND + n0g + n];
  }
  __syncthreads();

  const int wave = t >> 5, lane = t & 31;
  const int half = lane >> 4, r = lane & 15;
  const int rw = row0 + wave * 16;

  v8f am[2], al[2];
#pragma unroll
  for (int d = 0; d < 2; ++d)
#pragma unroll
    for (int v = 0; v < 8; ++v) { am[d][v] = 0.0f; al[d][v] = 0.0f; }

  for (int kk = 0; kk < 8; ++kk) {
    const v16h a = load_a_f16(hbuf + rw * NENC + kk * 32, NENC, lane);
#pragma unroll
    for (int d = 0; d < 2; ++d) {
      v16h b = load_b_T(WmuT, LDT, d * 16, kk * 32, lane);
      am[d] = __builtin_amdgcn_wmma_f32_16x16x32_f16(false, a, false, b,
                                                     (short)0, am[d], false, false);
      b = load_b_T(WlvT, LDT, d * 16, kk * 32, lane);
      al[d] = __builtin_amdgcn_wmma_f32_16x16x32_f16(false, a, false, b,
                                                     (short)0, al[d], false, false);
    }
  }
#pragma unroll
  for (int d = 0; d < 2; ++d) {
    const int n = n0g + d * 16 + r;
    const float bm = bmu[n], bl = blv[n];
#pragma unroll
    for (int v = 0; v < 8; ++v) {
      const int m = rw + v + 8 * half;
      const float muv = am[d][v] + bm;
      const float lvv = al[d][v] + bl;
      mu_o[m * ND + n] = muv;
      lv_o[m * ND + n] = lvv;
      z_o[m * ND + n]  = fmaf(eps[m * ND + n], __expf(0.5f * lvv), muv);
    }
  }
}

// ---------------------------------------------------------------------------
// Kernel 6: fused decoder — one block per batch element, 8 waves.
//   Bz = tanh(z*W_emb + b_emb)            (LDS, K-major, float4 weight reads)
//   T1 = M @ Bz          (WMMA, K=256)    (LDS, overlaid on Bz slab)
//   H1 = relu(T1 @ W_g1 + b_g1)           (VALU, Wg1 broadcast from LDS)
//   T2 = M @ H1          (WMMA, K=256)    (LDS, overlaid slab)
//   x_hat = T2 @ W_g2 + b_g2
// Static LDS < 64 KB via phase-overlaid regions with conflict-spreading pads.
// Each wave owns two 16-row strips of M so every B fragment feeds 2 WMMAs.
// ---------------------------------------------------------------------------
#define REG0   33792                         // BzT (64x264 h) = T1 (256x66 h) = T2 (256x33 f)
#define H1OFF  (REG0)                        // H1T: 32 x 264 halves = 16896 B
#define WGOFF  (REG0 + 16896)                // Wg1s(2048f) + bg1s(32f) + wg2s(32f) + bg2s(1f)
#define SMEMSZ (WGOFF + (2048 + 32 + 32 + 1) * 4)

__global__ __launch_bounds__(256) void k_dec(const float* __restrict__ z,
                                             const _Float16* __restrict__ Mh,
                                             const float* __restrict__ Wemb,
                                             const float* __restrict__ bemb,
                                             const float* __restrict__ Wg1,
                                             const float* __restrict__ bg1,
                                             const float* __restrict__ Wg2,
                                             const float* __restrict__ bg2,
                                             float* __restrict__ xhat) {
  __shared__ __align__(16) char smem[SMEMSZ];
  _Float16* BzT  = (_Float16*)smem;                 // [64][LDT] halves
  _Float16* T1   = (_Float16*)smem;                 // [256][66] halves (overlay)
  float*    T2f  = (float*)smem;                    // [256][33] f32   (overlay)
  _Float16* H1T  = (_Float16*)(smem + H1OFF);       // [32][LDT] halves
  float*    Wg1s = (float*)(smem + WGOFF);
  float*    bg1s = Wg1s + 2048;
  float*    wg2s = bg1s + 32;
  float*    bg2s = wg2s + 32;

  const int t  = threadIdx.x;
  const int bb = blockIdx.x;
  const int wave = t >> 5, lane = t & 31;
  const int half = lane >> 4, r = lane & 15;
  const int it0 = wave * 2;                 // M strip pair owned by this wave

  // ---- prefetch M into near caches (1024 x 128B lines, 4 per thread) ----
#pragma unroll
  for (int q = 0; q < 4; ++q)
    __builtin_prefetch(Mh + (t + q * 256) * 64, 0, 3);

  // ---- phase 0: stage small dense weights -------------------------------
  for (int idx = t; idx < 2048; idx += 256) Wg1s[idx] = Wg1[idx];
  if (t < 32) { bg1s[t] = bg1[t]; wg2s[t] = Wg2[t]; }
  if (t == 0) bg2s[0] = bg2[0];

  // ---- phase 1: per-node embedding Bz (K-major for WMMA B operand) ------
  {
    const float zv = z[bb * ND + t];
    const float4* w4 = (const float4*)(Wemb + t * NDEC);
    const float4* c4 = (const float4*)(bemb + t * NDEC);
#pragma unroll 4
    for (int q = 0; q < NDEC / 4; ++q) {
      const float4 w = w4[q], c = c4[q];
      BzT[(4 * q + 0) * LDT + t] = (_Float16)fast_tanhf(fmaf(zv, w.x, c.x));
      BzT[(4 * q + 1) * LDT + t] = (_Float16)fast_tanhf(fmaf(zv, w.y, c.y));
      BzT[(4 * q + 2) * LDT + t] = (_Float16)fast_tanhf(fmaf(zv, w.z, c.z));
      BzT[(4 * q + 3) * LDT + t] = (_Float16)fast_tanhf(fmaf(zv, w.w, c.w));
    }
  }
  __syncthreads();

  // ---- phase 2a: T1 = M @ Bz  (accumulators stay in VGPRs) --------------
  v8f acc[2][4];
#pragma unroll
  for (int ii = 0; ii < 2; ++ii)
#pragma unroll
    for (int d = 0; d < 4; ++d)
#pragma unroll
      for (int v = 0; v < 8; ++v) acc[ii][d][v] = 0.0f;

  for (int kk = 0; kk < 8; ++kk) {
    const v16h a0 = load_a_f16(Mh + (it0 + 0) * 16 * ND + kk * 32, ND, lane);
    const v16h a1 = load_a_f16(Mh + (it0 + 1) * 16 * ND + kk * 32, ND, lane);
#pragma unroll
    for (int d = 0; d < 4; ++d) {
      const v16h b = load_b_T(BzT, LDT, d * 16, kk * 32, lane);
      acc[0][d] = __builtin_amdgcn_wmma_f32_16x16x32_f16(false, a0, false, b,
                                                         (short)0, acc[0][d], false, false);
      acc[1][d] = __builtin_amdgcn_wmma_f32_16x16x32_f16(false, a1, false, b,
                                                         (short)0, acc[1][d], false, false);
    }
  }
  __syncthreads();              // all Bz reads done -> slab reusable as T1

  // ---- phase 2b: spill T1 (f16, padded stride 66) -----------------------
#pragma unroll
  for (int ii = 0; ii < 2; ++ii) {
    const int it = it0 + ii;
#pragma unroll
    for (int d = 0; d < 4; ++d)
#pragma unroll
      for (int v = 0; v < 8; ++v)
        T1[(it * 16 + v + 8 * half) * 66 + d * 16 + r] = (_Float16)acc[ii][d][v];
  }
  __syncthreads();

  // ---- phase 3: H1 = relu(T1 @ W_g1 + b_g1), K-major in LDS -------------
  {
    float s[32];
#pragma unroll
    for (int c = 0; c < 32; ++c) s[c] = bg1s[c];
    for (int d = 0; d < NDEC; ++d) {
      const float t1d = (float)T1[t * 66 + d];
#pragma unroll
      for (int c = 0; c < 32; ++c) s[c] = fmaf(t1d, Wg1s[d * 32 + c], s[c]);
    }
#pragma unroll
    for (int c = 0; c < 32; ++c) H1T[c * LDT + t] = (_Float16)fmaxf(s[c], 0.0f);
  }
  __syncthreads();

  // ---- phase 4: T2 = M @ H1 ---------------------------------------------
  v8f acc2[2][2];
#pragma unroll
  for (int ii = 0; ii < 2; ++ii)
#pragma unroll
    for (int d = 0; d < 2; ++d)
#pragma unroll
      for (int v = 0; v < 8; ++v) acc2[ii][d][v] = 0.0f;

  for (int kk = 0; kk < 8; ++kk) {
    const v16h a0 = load_a_f16(Mh + (it0 + 0) * 16 * ND + kk * 32, ND, lane);
    const v16h a1 = load_a_f16(Mh + (it0 + 1) * 16 * ND + kk * 32, ND, lane);
#pragma unroll
    for (int d = 0; d < 2; ++d) {
      const v16h b = load_b_T(H1T, LDT, d * 16, kk * 32, lane);
      acc2[0][d] = __builtin_amdgcn_wmma_f32_16x16x32_f16(false, a0, false, b,
                                                          (short)0, acc2[0][d], false, false);
      acc2[1][d] = __builtin_amdgcn_wmma_f32_16x16x32_f16(false, a1, false, b,
                                                          (short)0, acc2[1][d], false, false);
    }
  }
  // T1 reads finished before the phase-3 barrier; safe to overlay T2 now.
#pragma unroll
  for (int ii = 0; ii < 2; ++ii) {
    const int it = it0 + ii;
#pragma unroll
    for (int d = 0; d < 2; ++d)
#pragma unroll
      for (int v = 0; v < 8; ++v)
        T2f[(it * 16 + v + 8 * half) * 33 + d * 16 + r] = acc2[ii][d][v];
  }
  __syncthreads();

  // ---- phase 5: x_hat = T2 @ W_g2 + b_g2 --------------------------------
  {
    float sx = bg2s[0];
#pragma unroll
    for (int c = 0; c < 32; ++c) sx = fmaf(T2f[t * 33 + c], wg2s[c], sx);
    xhat[bb * ND + t] = sx;
  }
}

// ---------------------------------------------------------------------------
// Host launcher
// ---------------------------------------------------------------------------
extern "C" void kernel_launch(void* const* d_in, const int* in_sizes, int n_in,
                              void* d_out, int out_size, void* d_ws, size_t ws_size,
                              hipStream_t stream) {
  (void)in_sizes; (void)n_in; (void)out_size; (void)ws_size;

  const float* x    = (const float*)d_in[0];
  const float* eps  = (const float*)d_in[1];
  const float* Wenc = (const float*)d_in[2];
  const float* benc = (const float*)d_in[3];
  const float* Wmu  = (const float*)d_in[4];
  const float* bmu  = (const float*)d_in[5];
  const float* Wlv  = (const float*)d_in[6];
  const float* blv  = (const float*)d_in[7];
  const float* gp   = (const float*)d_in[8];
  const float* Wemb = (const float*)d_in[9];
  const float* bemb = (const float*)d_in[10];
  const float* Wg1  = (const float*)d_in[11];
  const float* bg1  = (const float*)d_in[12];
  const float* Wg2  = (const float*)d_in[13];
  const float* bg2  = (const float*)d_in[14];
  const int*   iter = (const int*)d_in[15];

  // d_out: x_hat [8192*256] | adj [256*256] | mu [8192*256] | logvar [8192*256]
  float* out    = (float*)d_out;
  float* xhat_o = out;
  float* adj_o  = out + NB * ND;
  float* mu_o   = adj_o + ND * ND;
  float* lv_o   = mu_o + NB * ND;

  // workspace layout
  char* ws = (char*)d_ws;
  float*     adj_ws = (float*)ws;                          // 256 KB
  _Float16*  Mh     = (_Float16*)(ws + 262144);            // 128 KB
  float*     degs   = (float*)(ws + 393216);               // 2 KB
  _Float16*  h16    = (_Float16*)(ws + 395264);            // 4 MB
  float*     zbuf   = (float*)(ws + 395264 + 4194304);     // 8 MB

  k_adj <<<ND, 256, 0, stream>>>(gp, iter, adj_o, adj_ws);
  k_deg <<<2, 256, 0, stream>>>(adj_ws, degs);
  k_M   <<<ND, 256, 0, stream>>>(adj_ws, degs, Mh);
  k_enc1<<<dim3(NB / 256, 4), 256, 0, stream>>>(x, Wenc, benc, h16);
  k_enc2<<<dim3(NB / 128, 8), 256, 0, stream>>>(h16, Wmu, bmu, Wlv, blv, eps,
                                                mu_o, lv_o, zbuf);
  k_dec <<<NB, 256, 0, stream>>>(zbuf, Mh, Wemb, bemb, Wg1, bg1, Wg2, bg2, xhat_o);
}
